// GDN1D_13211319403219
// MI455X (gfx1250) — matrix operand order repaired and checked by reading, not compiled, not run on hardware
//
#include <hip/hip_runtime.h>

#define F 256
#define ROWS_PER_BLOCK 128
#define LDSTRIDE 264   // padded LDS row stride in f16 elements (16B pad -> 4-bank lane skew)
#define BATCH 262144

typedef __attribute__((ext_vector_type(16))) _Float16 v16h;
typedef __attribute__((ext_vector_type(8)))  _Float16 v8h;
typedef __attribute__((ext_vector_type(4)))  _Float16 v4h;
typedef __attribute__((ext_vector_type(8)))  float    v8f;

// Prep: gammaT[n][k] = (f16) gamma[k][n]  (one-time 128KB transpose+convert into ws)
__global__ void gdn_prep_kernel(const float* __restrict__ gamma,
                                _Float16* __restrict__ gT) {
    const int n = blockIdx.x;   // 256 blocks
    const int k = threadIdx.x;  // 256 threads
    gT[n * F + k] = (_Float16)gamma[k * F + n];
}

__global__ __launch_bounds__(256)
void gdn_main_kernel(const float* __restrict__ x,
                     const float* __restrict__ beta,
                     const _Float16* __restrict__ gT,   // f16 gamma^T (may be null)
                     const float* __restrict__ gammaf,  // fp32 gamma fallback
                     float* __restrict__ y) {
    __shared__ alignas(16) _Float16 s_gT[F * LDSTRIDE];               // 135168 B
    __shared__ alignas(16) _Float16 s_xsq[ROWS_PER_BLOCK * LDSTRIDE]; // 67584 B
    __shared__ float s_beta[F];                                       // 1024 B

    const int tid = threadIdx.x;
    const size_t row0 = (size_t)blockIdx.x * ROWS_PER_BLOCK;

    if (tid < F) {
        float b = beta[tid];
        s_beta[tid] = b > 1e-6f ? b : 1e-6f;
    }

    // ---- stage gamma^T (f16) into LDS ----
    if (gT != nullptr) {
        const v8h* src = (const v8h*)gT;
        for (int i = tid; i < F * F / 8; i += 256) {
            const int e = i * 8;
            const int r = e >> 8;
            const int c = e & (F - 1);
            *(v8h*)&s_gT[r * LDSTRIDE + c] = src[i];
        }
    } else {
        // fallback: convert directly from fp32 gamma (slow, only if ws too small)
        for (int i = tid; i < F * F; i += 256) {
            const int r = i >> 8;        // n
            const int c = i & (F - 1);   // k
            s_gT[r * LDSTRIDE + c] = (_Float16)gammaf[c * F + r];
        }
    }

    // ---- stage x tile: coalesced fp32 load, square, f16 into LDS ----
    {
        const float4* xv = (const float4*)(x + row0 * F);
        for (int i = tid; i < ROWS_PER_BLOCK * F / 4; i += 256) {
            const float4 v = xv[i];
            const int e = i * 4;
            const int r = e >> 8;
            const int c = e & (F - 1);
            v4h q;
            q[0] = (_Float16)(v.x * v.x);
            q[1] = (_Float16)(v.y * v.y);
            q[2] = (_Float16)(v.z * v.z);
            q[3] = (_Float16)(v.w * v.w);
            *(v4h*)&s_xsq[r * LDSTRIDE + c] = q;
        }
    }
    __syncthreads();

    const int wave = tid >> 5;
    const int lane = tid & 31;
    const int l16  = lane & 15;
    const int hi   = lane >> 4;

    // ---- A fragments: row m = wave*16 + (lane&15); K split {c..c+7, c+16..c+23}, c = 32s + 8*hi
    const int mrow = wave * 16 + l16;
    v16h afrag[8];
#pragma unroll
    for (int s = 0; s < 8; ++s) {
        const int c = 32 * s + 8 * hi;
        const v8h lo  = *(const v8h*)&s_xsq[mrow * LDSTRIDE + c];
        const v8h hi8 = *(const v8h*)&s_xsq[mrow * LDSTRIDE + c + 16];
#pragma unroll
        for (int t = 0; t < 8; ++t) { afrag[s][t] = lo[t]; afrag[s][t + 8] = hi8[t]; }
    }

    // D rows handled by this lane: rowbase + r (r = 0..7)
    const int rowbase = wave * 16 + 8 * hi;

#pragma unroll
    for (int j = 0; j < 16; ++j) {
        v8f acc = {0.f, 0.f, 0.f, 0.f, 0.f, 0.f, 0.f, 0.f};
        const int n = j * 16 + l16;
#pragma unroll
        for (int s = 0; s < 8; ++s) {
            // B layout: K split {c..c+7, c+16..c+23}, c = 32s + 8*hi (matches A)
            const int cb = 32 * s + 8 * hi;
            const v8h b0 = *(const v8h*)&s_gT[n * LDSTRIDE + cb];
            const v8h b1 = *(const v8h*)&s_gT[n * LDSTRIDE + cb + 16];
            v16h b;
#pragma unroll
            for (int t = 0; t < 8; ++t) { b[t] = b0[t]; b[t + 8] = b1[t]; }
            acc = __builtin_amdgcn_wmma_f32_16x16x32_f16(
                false, afrag[s], false, b, (short)0, acc, false, false);
        }
        // epilogue: y[m][n] = x[m][n] * rsqrt(acc + beta_c[n])
        const float bc = s_beta[n];
        const float* xp = x + (row0 + rowbase) * F + n;
        float*       yp = y + (row0 + rowbase) * F + n;
#pragma unroll
        for (int r = 0; r < 8; ++r) {
            const float xv  = xp[(size_t)r * F];
            const float inv = rsqrtf(acc[r] + bc);
            yp[(size_t)r * F] = xv * inv;
        }
    }
}

extern "C" void kernel_launch(void* const* d_in, const int* in_sizes, int n_in,
                              void* d_out, int out_size, void* d_ws, size_t ws_size,
                              hipStream_t stream) {
    const float* x     = (const float*)d_in[0];
    const float* beta  = (const float*)d_in[1];
    const float* gamma = (const float*)d_in[2];
    float* y = (float*)d_out;

    const size_t gT_bytes = (size_t)F * F * sizeof(_Float16);  // 128 KB
    _Float16* gT = nullptr;
    if (ws_size >= gT_bytes && d_ws != nullptr) {
        gT = (_Float16*)d_ws;
        gdn_prep_kernel<<<F, F, 0, stream>>>(gamma, gT);
    }

    const int blocks = BATCH / ROWS_PER_BLOCK;  // 2048
    gdn_main_kernel<<<blocks, 256, 0, stream>>>(x, beta, gT, gamma, y);
}
